// MultiHeadAttention_31542239822105
// MI455X (gfx1250) — matrix operand order repaired and checked by reading, not compiled, not run on hardware
//
#include <hip/hip_runtime.h>
#include <hip/hip_bf16.h>
#include <hip/hip_fp16.h>

#define B_  64
#define S_  577
#define D_  768
#define H_  12
#define DH_ 64
#define SP_ 608           // S padded to multiple of 32 (38 tiles of 16)
#define NT_ (SP_ / 16)    // 38 tiles
#define NKP_ (SP_ / 32)   // 19 k-pairs

typedef __attribute__((ext_vector_type(16))) _Float16 v16h;
typedef __attribute__((ext_vector_type(8)))  _Float16 v8h;
typedef __attribute__((ext_vector_type(8)))  float    v8f;

#define ZERO8 {0.f,0.f,0.f,0.f,0.f,0.f,0.f,0.f}

// ---- WMMA fragment loaders (CDNA5 wave32 layouts, CDNA5 ISA) ----

// A 16x32 f16: lane%16 = M row; lanes<16 hold K 0..7,16..23; lanes>=16 hold K 8..15,24..31
static __device__ inline v16h load_a_f16(const _Float16* tile, int ld, int lane) {
    const _Float16* p = tile + (lane & 15) * ld + ((lane >> 4) & 1) * 8;
    v8h lo = *(const v8h*)p;
    v8h hi = *(const v8h*)(p + 16);
    return __builtin_shufflevector(lo, hi, 0,1,2,3,4,5,6,7,8,9,10,11,12,13,14,15);
}

// B 32x16 f16 from a [N][K] (transposed, K contiguous) matrix:
// lane%16 = N column; lanes<16 hold K 0..7,16..23; lanes>=16 hold K 8..15,24..31
static __device__ inline v16h load_b_f16(const _Float16* tile, int ld, int lane) {
    const _Float16* p = tile + (lane & 15) * ld + ((lane >> 4) & 1) * 8;
    v8h lo = *(const v8h*)p;
    v8h hi = *(const v8h*)(p + 16);
    return __builtin_shufflevector(lo, hi, 0,1,2,3,4,5,6,7,8,9,10,11,12,13,14,15);
}

static __device__ inline v8f wmma_f16(v16h a, v16h b, v8f c) {
    return __builtin_amdgcn_wmma_f32_16x16x32_f16(false, a, false, b, (short)0, c, false, false);
}

// ---------------- Kernel 1a: weight prep (fp32 -> f16, transpose to [n][k]) ---
__global__ __launch_bounds__(256) void mha_prep_weights(
        const float* Wq, const float* Wk, const float* Wv, const float* Wo,
        _Float16* Wt, _Float16* Wot) {
    const long NQKV = 3L * H_ * D_ * DH_;
    const long NO   = (long)D_ * D_;
    const long total = NQKV + NO;
    for (long i = (long)blockIdx.x * blockDim.x + threadIdx.x; i < total;
         i += (long)gridDim.x * blockDim.x) {
        if (i < NQKV) {
            int e = (int)(i % DH_);
            int d = (int)((i / DH_) % D_);
            int h = (int)((i / ((long)DH_ * D_)) % H_);
            int p = (int)(i / ((long)DH_ * D_ * H_));
            const float* W = (p == 0) ? Wq : (p == 1) ? Wk : Wv;
            float v = W[((long)h * D_ + d) * DH_ + e];
            Wt[(((long)p * H_ + h) * DH_ + e) * D_ + d] = (_Float16)v;
        } else {
            long j = i - NQKV;
            int k = (int)(j % D_);
            int n = (int)(j / D_);
            Wot[(long)n * D_ + k] = (_Float16)Wo[(long)k * D_ + n];
        }
    }
}

// ---------------- Kernel 1b: x fp32 -> f16 (read once, reused 36x) ------------
__global__ __launch_bounds__(256) void mha_prep_x(const float* x, _Float16* xh) {
    const long N = (long)B_ * S_ * D_;
    for (long i = (long)blockIdx.x * blockDim.x + threadIdx.x; i < N;
         i += (long)gridDim.x * blockDim.x) {
        xh[i] = (_Float16)x[i];
    }
}

// ---------------- Kernel 2: QKV projection GEMM -------------------------------
// grid.x = 577 (64-row tiles of M=B*S), grid.y = 36 (proj*12+h), 8 waves/block
__global__ __launch_bounds__(256) void mha_qkv_proj(
        const _Float16* xh, const _Float16* Wt,
        const float* bq, const float* bk, const float* bv, _Float16* qkv) {
    const int lane = threadIdx.x & 31;
    const int wave = threadIdx.x >> 5;
    const int g    = blockIdx.y;
    const int proj = g / H_;
    const int h    = g % H_;
    const int row0 = blockIdx.x * 64 + (wave >> 1) * 16;
    const int ntb  = (wave & 1) * 2;            // n-tiles ntb, ntb+1
    const _Float16* Wb = Wt + (long)g * DH_ * D_;
    const float* bias = ((proj == 0) ? bq : (proj == 1) ? bk : bv) + h * DH_;

    v8f c0 = ZERO8, c1 = ZERO8;
    for (int kk = 0; kk < D_; kk += 32) {
        v16h a  = load_a_f16(xh + (long)row0 * D_ + kk, D_, lane);
        v16h b0 = load_b_f16(Wb + (long)(ntb * 16) * D_ + kk, D_, lane);
        v16h b1 = load_b_f16(Wb + (long)(ntb * 16 + 16) * D_ + kk, D_, lane);
        c0 = wmma_f16(a, b0, c0);
        c1 = wmma_f16(a, b1, c1);
    }
    const long pstride = (long)B_ * H_ * S_ * DH_;
    _Float16* out = qkv + (long)proj * pstride;
    const int mofs = ((lane >> 4) & 1) * 8;
    const int n0 = ntb * 16 + (lane & 15);
    const float bb0 = bias[n0], bb1 = bias[n0 + 16];
#pragma unroll
    for (int r = 0; r < 8; ++r) {
        int row = row0 + r + mofs;
        int b = row / S_;
        int s = row - b * S_;
        long base = (((long)b * H_ + h) * S_ + s) * DH_;
        out[base + n0]      = (_Float16)(c0[r] + bb0);
        out[base + n0 + 16] = (_Float16)(c1[r] + bb1);
    }
}

// ---------------- Kernel 3: attention core (one workgroup per (b,h)) ----------
// Q,K,Vt resident in LDS (CDNA5: 320KB/WGP). Softmax over the QUERY axis
// (faithful to reference): pass1 = per-key-column (max,sumexp), pass2 = P@V.
__global__ __launch_bounds__(256) void mha_attention(const _Float16* qkv, _Float16* ctx) {
    extern __shared__ char smem[];
    _Float16* Qs = (_Float16*)smem;              // [SP_][DH_]
    _Float16* Ks = Qs + SP_ * DH_;               // [SP_][DH_]
    _Float16* Vt = Ks + SP_ * DH_;               // [DH_][SP_] (transposed)
    float* stm = (float*)(Vt + DH_ * SP_);       // per-column max   [SP_]
    float* srz = stm + SP_;                      // per-column 1/Z   [SP_]
    _Float16* Pbuf = (_Float16*)(srz + SP_);     // 8 waves * 16*32 f16

    const int tid  = threadIdx.x;
    const int lane = tid & 31;
    const int wave = tid >> 5;
    const int bh   = blockIdx.x;                 // b*H + h
    const long pstride = (long)B_ * H_ * S_ * DH_;
    const _Float16* qg = qkv + (long)bh * S_ * DH_;
    const _Float16* kg = qg + pstride;
    const _Float16* vg = qg + 2 * pstride;

    // ---- phase 0: stage Q, K into LDS via CDNA5 async-to-LDS DMA -------------
    // (CDNA5 ISA: LDS address taken from a VGPR, ASYNCcnt-tracked)
    typedef __attribute__((address_space(3))) char lds3_char;
    const unsigned qs_lds = (unsigned)(size_t)(lds3_char*)(void*)Qs;
    const unsigned ks_lds = (unsigned)(size_t)(lds3_char*)(void*)Ks;

    // zero-pad tail rows [S_, SP_) with ordinary DS stores first
    for (int i = tid; i < (SP_ - S_) * (DH_ / 8); i += blockDim.x) {
        int row = S_ + (i >> 3), wi = i & 7;
        uint4 z; z.x = z.y = z.z = z.w = 0u;
        *(uint4*)(Qs + row * DH_ + wi * 8) = z;
        *(uint4*)(Ks + row * DH_ + wi * 8) = z;
    }
    // async DMA the valid region, 16B per lane per issue
    {
        const int nchunk = S_ * (DH_ / 8);       // 4616 x 16B chunks
        for (int i = tid; i < nchunk; i += blockDim.x) {
            const unsigned lofs = (unsigned)(i * 16);
            const char* gq = (const char*)qg + (long)i * 16;
            const char* gk = (const char*)kg + (long)i * 16;
            asm volatile("global_load_async_to_lds_b128 %0, %1, off"
                         :: "v"(qs_lds + lofs), "v"(gq) : "memory");
            asm volatile("global_load_async_to_lds_b128 %0, %1, off"
                         :: "v"(ks_lds + lofs), "v"(gk) : "memory");
        }
    }
    // V needs a transpose, so it goes through VGPRs
    for (int i = tid; i < SP_ * DH_; i += blockDim.x) {
        int k = i >> 6, e = i & 63;
        _Float16 v = (k < S_) ? vg[(long)k * DH_ + e] : (_Float16)0.0f;
        Vt[e * SP_ + k] = v;
    }
    asm volatile("s_wait_asynccnt 0x0" ::: "memory");
    __syncthreads();

    const float scale = 0.125f;                  // 1/sqrt(64)
    const float NEGINF = -__builtin_inff();
    const int mofs = ((lane >> 4) & 1) * 8;

    // ---- phase 1: column stats; D-layout => each lane owns one column -------
    for (int kt = wave; kt < NT_; kt += 8) {
        v16h bk0 = load_b_f16(Ks + kt * 16 * DH_ + 0,  DH_, lane);
        v16h bk1 = load_b_f16(Ks + kt * 16 * DH_ + 32, DH_, lane);
        float m = NEGINF, Z = 0.0f;
        for (int qt = 0; qt < NT_; ++qt) {
            v16h a0 = load_a_f16(Qs + qt * 16 * DH_ + 0,  DH_, lane);
            v16h a1 = load_a_f16(Qs + qt * 16 * DH_ + 32, DH_, lane);
            v8f c = ZERO8;
            c = wmma_f16(a0, bk0, c);
            c = wmma_f16(a1, bk1, c);
            float sv[8];
            float tmax = NEGINF;
#pragma unroll
            for (int r = 0; r < 8; ++r) {
                int q = qt * 16 + r + mofs;
                sv[r] = (q < S_) ? c[r] * scale : NEGINF;
                tmax = fmaxf(tmax, sv[r]);
            }
            tmax = fmaxf(tmax, __shfl_xor(tmax, 16, 32));
            float mn = fmaxf(m, tmax);
            float p = 0.0f;
#pragma unroll
            for (int r = 0; r < 8; ++r) p += __expf(sv[r] - mn);
            p += __shfl_xor(p, 16, 32);
            Z = Z * __expf(m - mn) + p;
            m = mn;
        }
        if (lane < 16) {
            stm[kt * 16 + lane] = m;
            srz[kt * 16 + lane] = 1.0f / Z;
        }
    }
    __syncthreads();

    // ---- phase 2: ctx = P @ V, recomputing scores; P re-tiled via LDS --------
    _Float16* pb = Pbuf + wave * (16 * 32);
    const int b = bh / H_, h = bh % H_;
    for (int mt = wave; mt < NT_; mt += 8) {
        v16h aq0 = load_a_f16(Qs + mt * 16 * DH_ + 0,  DH_, lane);
        v16h aq1 = load_a_f16(Qs + mt * 16 * DH_ + 32, DH_, lane);
        v8f acc0 = ZERO8, acc1 = ZERO8, acc2 = ZERO8, acc3 = ZERO8;
        for (int kp = 0; kp < NKP_; ++kp) {
#pragma unroll
            for (int sub = 0; sub < 2; ++sub) {
                int kt = kp * 2 + sub;
                v16h bk0 = load_b_f16(Ks + kt * 16 * DH_ + 0,  DH_, lane);
                v16h bk1 = load_b_f16(Ks + kt * 16 * DH_ + 32, DH_, lane);
                v8f c = ZERO8;
                c = wmma_f16(aq0, bk0, c);
                c = wmma_f16(aq1, bk1, c);
                int kg2 = kt * 16 + (lane & 15);
                float mk = stm[kg2], rz = srz[kg2];
#pragma unroll
                for (int r = 0; r < 8; ++r) {
                    float pv = __expf(c[r] * scale - mk) * rz;
                    pv = (kg2 < S_) ? pv : 0.0f;
                    pb[(r + mofs) * 32 + sub * 16 + (lane & 15)] = (_Float16)pv;
                }
            }
            // same-wave LDS RAW is in-order in HW; stop compiler reordering only
            asm volatile("" ::: "memory");
            v16h ap = load_a_f16(pb, 32, lane);
            acc0 = wmma_f16(ap, load_b_f16(Vt + 0 * 16 * SP_ + kp * 32, SP_, lane), acc0);
            acc1 = wmma_f16(ap, load_b_f16(Vt + 1 * 16 * SP_ + kp * 32, SP_, lane), acc1);
            acc2 = wmma_f16(ap, load_b_f16(Vt + 2 * 16 * SP_ + kp * 32, SP_, lane), acc2);
            acc3 = wmma_f16(ap, load_b_f16(Vt + 3 * 16 * SP_ + kp * 32, SP_, lane), acc3);
            asm volatile("" ::: "memory");
        }
#pragma unroll
        for (int r = 0; r < 8; ++r) {
            int q = mt * 16 + r + mofs;
            if (q < S_) {
                long base = ((long)(b * S_ + q)) * D_ + h * DH_ + (lane & 15);
                ctx[base +  0] = (_Float16)acc0[r];
                ctx[base + 16] = (_Float16)acc1[r];
                ctx[base + 32] = (_Float16)acc2[r];
                ctx[base + 48] = (_Float16)acc3[r];
            }
        }
    }
}

// ---------------- Kernel 4: output projection GEMM + bias ---------------------
__global__ __launch_bounds__(256) void mha_out_proj(
        const _Float16* ctxh, const _Float16* Wot, const float* bo, float* out) {
    const int lane = threadIdx.x & 31;
    const int wave = threadIdx.x >> 5;
    const int row0 = blockIdx.x * 64 + (wave >> 1) * 16;
    const int n0   = blockIdx.y * 64 + (wave & 1) * 32;

    v8f c0 = ZERO8, c1 = ZERO8;
    for (int kk = 0; kk < D_; kk += 32) {
        v16h a  = load_a_f16(ctxh + (long)row0 * D_ + kk, D_, lane);
        v16h b0 = load_b_f16(Wot + (long)n0 * D_ + kk, D_, lane);
        v16h b1 = load_b_f16(Wot + (long)(n0 + 16) * D_ + kk, D_, lane);
        c0 = wmma_f16(a, b0, c0);
        c1 = wmma_f16(a, b1, c1);
    }
    const int mofs = ((lane >> 4) & 1) * 8;
    const int nc = n0 + (lane & 15);
    const float bb0 = bo[nc], bb1 = bo[nc + 16];
#pragma unroll
    for (int r = 0; r < 8; ++r) {
        int row = row0 + r + mofs;
        out[(long)row * D_ + nc]      = c0[r] + bb0;
        out[(long)row * D_ + nc + 16] = c1[r] + bb1;
    }
}

// ---------------- launcher ----------------------------------------------------
extern "C" void kernel_launch(void* const* d_in, const int* in_sizes, int n_in,
                              void* d_out, int out_size, void* d_ws, size_t ws_size,
                              hipStream_t stream) {
    const float* x  = (const float*)d_in[0];
    const float* Wq = (const float*)d_in[1];
    const float* bq = (const float*)d_in[2];
    const float* Wk = (const float*)d_in[3];
    const float* bk = (const float*)d_in[4];
    const float* Wv = (const float*)d_in[5];
    const float* bv = (const float*)d_in[6];
    const float* Wo = (const float*)d_in[7];
    const float* bo = (const float*)d_in[8];
    float* out = (float*)d_out;

    _Float16* Wt  = (_Float16*)d_ws;                 // 3*H*DH*D   f16
    _Float16* Wot = Wt + 3L * H_ * DH_ * D_;         // D*D        f16
    _Float16* qkv = Wot + (long)D_ * D_;             // 3*B*H*S*DH f16
    _Float16* ctx = qkv + 3L * B_ * H_ * S_ * DH_;   // B*S*D      f16
    _Float16* xh  = ctx + (long)B_ * S_ * D_;        // B*S*D      f16

    mha_prep_weights<<<dim3(512), dim3(256), 0, stream>>>(Wq, Wk, Wv, Wo, Wt, Wot);
    mha_prep_x<<<dim3(2048), dim3(256), 0, stream>>>(x, xh);
    mha_qkv_proj<<<dim3(S_, 36), dim3(256), 0, stream>>>(xh, Wt, bq, bk, bv, qkv);

    size_t smem = (size_t)(3 * SP_ * DH_ + 8 * 16 * 32) * sizeof(_Float16)
                + (size_t)(2 * SP_) * sizeof(float);   // 246,528 B < 320KB WGP LDS
    mha_attention<<<dim3(B_ * H_), dim3(256), smem, stream>>>(qkv, ctx);

    mha_out_proj<<<dim3(S_, D_ / 64), dim3(256), 0, stream>>>(ctx, Wot, bo, out);
}
